// SAGE_5282809775001
// MI455X (gfx1250) — matrix operand order repaired and checked by reading, not compiled, not run on hardware
//
#include <hip/hip_runtime.h>

#define NN 50000
#define NE 600000

typedef __attribute__((ext_vector_type(16))) __bf16 v16bf;
typedef __attribute__((ext_vector_type(8)))  __bf16 v8bf;
typedef __attribute__((ext_vector_type(8)))  float  v8f;

__device__ __forceinline__ unsigned short f32_to_bf16_rne(float f) {
  unsigned int u = __float_as_uint(f);
  unsigned int lsb = (u >> 16) & 1u;
  u += 0x7fffu + lsb;
  return (unsigned short)(u >> 16);
}

__global__ void zero_f32_kernel(float* __restrict__ p, int n) {
  int i = blockIdx.x * blockDim.x + threadIdx.x;
  if (i < n) p[i] = 0.0f;
}

__global__ void cvt_f32_bf16_kernel(const float* __restrict__ in,
                                    unsigned short* __restrict__ out, int n) {
  int i = blockIdx.x * blockDim.x + threadIdx.x;
  if (i < n) out[i] = f32_to_bf16_rne(in[i]);
}

__global__ void degree_kernel(const int* __restrict__ dst, float* __restrict__ deg, int E) {
  int e = blockIdx.x * blockDim.x + threadIdx.x;
  if (e < E) atomicAdd(&deg[dst[e]], 1.0f);
}

// One thread handles 4 channels of one edge: float4 gather + 4 L2-resident atomics.
template<int C>
__global__ void scatter_add_kernel(const float* __restrict__ h, const int* __restrict__ src,
                                   const int* __restrict__ dst, float* __restrict__ agg, int E) {
  constexpr int CV = C / 4;
  int t = blockIdx.x * blockDim.x + threadIdx.x;
  if (t >= E * CV) return;
  int e = t / CV;
  int c = (t - e * CV) * 4;
  int s = src[e], d = dst[e];
  const float4 v = *(const float4*)(h + (size_t)s * C + c);
  float* a = agg + (size_t)d * C + c;
  atomicAdd(a + 0, v.x);
  atomicAdd(a + 1, v.y);
  atomicAdd(a + 2, v.z);
  atomicAdd(a + 3, v.w);
}

// agg / max(deg,1) fused with f32->bf16
template<int C>
__global__ void normalize_bf16_kernel(const float* __restrict__ agg, const float* __restrict__ deg,
                                      unsigned short* __restrict__ out, int n) {
  int i = blockIdx.x * blockDim.x + threadIdx.x;
  if (i >= n) return;
  int node = i / C;
  float inv = 1.0f / fmaxf(deg[node], 1.0f);
  out[i] = f32_to_bf16_rne(agg[i] * inv);
}

// A-fragment, contiguous K order: lanes 0-15 hold K {kt..kt+15},
// lanes 16-31 hold K {kt+16..kt+31}. Two contiguous 16B loads.
__device__ __forceinline__ v16bf load_a_tile(const unsigned short* p, int hi) {
  const __bf16* b = (const __bf16*)p;
  v8bf lo = *(const v8bf*)(b + 16 * hi);
  v8bf up = *(const v8bf*)(b + 16 * hi + 8);
  v16bf r;
#pragma unroll
  for (int i = 0; i < 8; ++i) { r[i] = lo[i]; r[i + 8] = up[i]; }
  return r;
}

// Fused dual GEMM: out[m,o] = act( aggb[m,:] @ Wl[o,:] + hb[m,:] @ Wr[o,:] )
// Block = 8 waves = 8 consecutive M-tiles sharing one 64-column n-block.
// The block stages its Wl/Wr slices (contiguous rows of row-major OxK) into LDS
// via async global->LDS copies (ASYNCcnt), then each wave runs the K-loop with
// 8 v_wmma_f32_16x16x32_bf16 per step, A reused 4x, B from LDS.
template<int K, int OPAD, int OSTORE, bool RELU, bool WBF16>
__global__ __launch_bounds__(256) void sage_gemm_wmma(
    const unsigned short* __restrict__ aggb, const unsigned short* __restrict__ hb,
    const unsigned short* __restrict__ Wl,   const unsigned short* __restrict__ Wr,
    float* __restrict__ out, unsigned short* __restrict__ hbout, int M) {
  constexpr int TILES_N = OPAD / 64;
  __shared__ alignas(64) unsigned short ldsW[2][64][K];

  const int nb = (int)(blockIdx.x % TILES_N);
  const int mg = (int)(blockIdx.x / TILES_N);
  const int n0 = nb * 64;

  // ---- async stage: Wl/Wr rows [n0, n0+64) are one contiguous 128*K-byte block each
  {
    const unsigned int t = threadIdx.x;                 // 0..255
    constexpr int UNITS = (64 * K * 2) / 16;            // 16B units per matrix slice
    const unsigned int lds0 = (unsigned int)(unsigned long long)&ldsW[0][0][0];
    const unsigned int lds1 = (unsigned int)(unsigned long long)&ldsW[1][0][0];
    const unsigned long long g0 = (unsigned long long)(Wl + (size_t)n0 * K);
    const unsigned long long g1 = (unsigned long long)(Wr + (size_t)n0 * K);
#pragma unroll
    for (int c = 0; c < UNITS / 256; ++c) {
      const unsigned int byteOff = (t + c * 256) * 16;
      unsigned int l0 = lds0 + byteOff;
      unsigned long long a0 = g0 + byteOff;
      asm volatile("global_load_async_to_lds_b128 %0, %1, off"
                   :: "v"(l0), "v"(a0) : "memory");
      unsigned int l1 = lds1 + byteOff;
      unsigned long long a1 = g1 + byteOff;
      asm volatile("global_load_async_to_lds_b128 %0, %1, off"
                   :: "v"(l1), "v"(a1) : "memory");
    }
    asm volatile("s_wait_asynccnt 0x0" ::: "memory");   // this wave's copies done
  }
  __syncthreads();                                      // all waves' copies visible

  const int mt = mg * 8 + (int)(threadIdx.x >> 5);
  if (mt >= (M >> 4)) return;                           // uniform per wave (EXEC stays full)
  const int lane = threadIdx.x & 31;
  const int hi   = lane >> 4;
  const int l16  = lane & 15;
  const int m0   = mt << 4;
  const unsigned short* aRow = aggb + (size_t)(m0 + l16) * K;
  const unsigned short* xRow = hb   + (size_t)(m0 + l16) * K;

  v8f acc[4];
#pragma unroll
  for (int nn = 0; nn < 4; ++nn)
#pragma unroll
    for (int r = 0; r < 8; ++r) acc[nn][r] = 0.0f;

#pragma unroll
  for (int kt = 0; kt < K; kt += 32) {
    v16bf aAgg = load_a_tile(aRow + kt, hi);
    v16bf aX   = load_a_tile(xRow + kt, hi);
#pragma unroll
    for (int nn = 0; nn < 4; ++nn) {
      const int ol = nn * 16 + l16;                     // B column == row of W slice
      v16bf bl = *(const v16bf*)&ldsW[0][ol][kt + 16 * hi];
      v16bf br = *(const v16bf*)&ldsW[1][ol][kt + 16 * hi];
      acc[nn] = __builtin_amdgcn_wmma_f32_16x16x32_bf16(
          false, aAgg, false, bl, (short)0, acc[nn], false, false);
      acc[nn] = __builtin_amdgcn_wmma_f32_16x16x32_bf16(
          false, aX,   false, br, (short)0, acc[nn], false, false);
    }
  }

  // C/D layout: VGPR r, lanes 0-15 -> M=m0+r, lanes 16-31 -> M=m0+8+r; N = l16
#pragma unroll
  for (int nn = 0; nn < 4; ++nn) {
    const int col = n0 + nn * 16 + l16;
    if (col < OSTORE) {                                 // compile-time foldable vs OSTORE
#pragma unroll
      for (int r = 0; r < 8; ++r) {
        float v = acc[nn][r];
        if (RELU) v = fmaxf(v, 0.0f);
        const size_t idx = (size_t)(m0 + 8 * hi + r) * OSTORE + col;
        out[idx] = v;
        if (WBF16) hbout[idx] = f32_to_bf16_rne(v);     // fused bf16 copy for next layer
      }
    }
  }
}

// One wave32 per row (O=47): shuffle-reduce max & sum, rewrite in place.
__global__ __launch_bounds__(256) void log_softmax_kernel(float* __restrict__ out, int N, int O) {
  const int wid = blockIdx.x * (blockDim.x >> 5) + (threadIdx.x >> 5);
  if (wid >= N) return;
  const int lane = threadIdx.x & 31;
  float* row = out + (size_t)wid * O;
  float v0 = (lane < O)      ? row[lane]      : -3.0e38f;
  float v1 = (lane + 32 < O) ? row[lane + 32] : -3.0e38f;
  float m = fmaxf(v0, v1);
#pragma unroll
  for (int off = 16; off > 0; off >>= 1) m = fmaxf(m, __shfl_xor(m, off, 32));
  float s = ((lane < O) ? expf(v0 - m) : 0.0f) + ((lane + 32 < O) ? expf(v1 - m) : 0.0f);
#pragma unroll
  for (int off = 16; off > 0; off >>= 1) s += __shfl_xor(s, off, 32);
  const float lse = logf(s) + m;
  if (lane < O)      row[lane]      = v0 - lse;
  if (lane + 32 < O) row[lane + 32] = v1 - lse;
}

extern "C" void kernel_launch(void* const* d_in, const int* in_sizes, int n_in,
                              void* d_out, int out_size, void* d_ws, size_t ws_size,
                              hipStream_t stream) {
  (void)in_sizes; (void)n_in; (void)out_size; (void)ws_size;
  constexpr int N = NN, E = NE;
  const float* x   = (const float*)d_in[0];
  const float* Wl1 = (const float*)d_in[1];   // 256x128
  const float* Wr1 = (const float*)d_in[2];   // 256x128
  const float* Wl2 = (const float*)d_in[3];   // 256x256
  const float* Wr2 = (const float*)d_in[4];   // 256x256
  const float* Wl3 = (const float*)d_in[5];   // 47x256
  const float* Wr3 = (const float*)d_in[6];   // 47x256
  const int* edge  = (const int*)d_in[7];     // int32 (JAX x64 off)
  const int* src = edge;
  const int* dst = edge + E;
  float* out = (float*)d_out;

  char* base = (char*)d_ws;
  size_t off = 0;
  auto alloc = [&](size_t bytes) -> void* {
    void* p = base + off;
    off = (off + bytes + 255) & ~(size_t)255;
    return p;
  };
  float* deg  = (float*)alloc((size_t)N * 4);
  float* agg  = (float*)alloc((size_t)N * 256 * 4);
  float* h    = (float*)alloc((size_t)N * 256 * 4);
  unsigned short* hbA  = (unsigned short*)alloc((size_t)N * 256 * 2);
  unsigned short* hbB  = (unsigned short*)alloc((size_t)N * 256 * 2);
  unsigned short* aggb = (unsigned short*)alloc((size_t)N * 256 * 2);
  unsigned short* wb1l = (unsigned short*)alloc(256 * 128 * 2);
  unsigned short* wb1r = (unsigned short*)alloc(256 * 128 * 2);
  unsigned short* wb2l = (unsigned short*)alloc(256 * 256 * 2);
  unsigned short* wb2r = (unsigned short*)alloc(256 * 256 * 2);
  unsigned short* wb3l = (unsigned short*)alloc(64 * 256 * 2);   // zero-padded 47 -> 64 rows
  unsigned short* wb3r = (unsigned short*)alloc(64 * 256 * 2);

  const int T = 256;
  auto nblk = [](long n, int t) { return (int)((n + t - 1) / t); };
  const int MGRP = (N / 16 + 7) / 8;          // 8 M-tiles per block

  // weights -> bf16 (per call; deterministic); layer-3 padded with zeros
  cvt_f32_bf16_kernel<<<nblk(256 * 128, T), T, 0, stream>>>(Wl1, wb1l, 256 * 128);
  cvt_f32_bf16_kernel<<<nblk(256 * 128, T), T, 0, stream>>>(Wr1, wb1r, 256 * 128);
  cvt_f32_bf16_kernel<<<nblk(256 * 256, T), T, 0, stream>>>(Wl2, wb2l, 256 * 256);
  cvt_f32_bf16_kernel<<<nblk(256 * 256, T), T, 0, stream>>>(Wr2, wb2r, 256 * 256);
  zero_f32_kernel<<<nblk(64 * 256 / 2, T), T, 0, stream>>>((float*)wb3l, 64 * 256 / 2);
  zero_f32_kernel<<<nblk(64 * 256 / 2, T), T, 0, stream>>>((float*)wb3r, 64 * 256 / 2);
  cvt_f32_bf16_kernel<<<nblk(47 * 256, T), T, 0, stream>>>(Wl3, wb3l, 47 * 256);
  cvt_f32_bf16_kernel<<<nblk(47 * 256, T), T, 0, stream>>>(Wr3, wb3r, 47 * 256);

  // degrees (shared by all layers)
  zero_f32_kernel<<<nblk(N, T), T, 0, stream>>>(deg, N);
  degree_kernel<<<nblk(E, T), T, 0, stream>>>(dst, deg, E);

  // root features -> bf16 once
  cvt_f32_bf16_kernel<<<nblk((long)N * 128, T), T, 0, stream>>>(x, hbA, N * 128);

  // ---- layer 1: 128 -> 256, ReLU; writes h (f32) + hbB (bf16) fused
  zero_f32_kernel<<<nblk((long)N * 128, T), T, 0, stream>>>(agg, N * 128);
  scatter_add_kernel<128><<<nblk((long)E * 32, T), T, 0, stream>>>(x, src, dst, agg, E);
  normalize_bf16_kernel<128><<<nblk((long)N * 128, T), T, 0, stream>>>(agg, deg, aggb, N * 128);
  sage_gemm_wmma<128, 256, 256, true, true><<<MGRP * 4, 256, 0, stream>>>(
      aggb, hbA, wb1l, wb1r, h, hbB, N);

  // ---- layer 2: 256 -> 256, ReLU; reads hbB, writes h (in place) + hbA
  zero_f32_kernel<<<nblk((long)N * 256, T), T, 0, stream>>>(agg, N * 256);
  scatter_add_kernel<256><<<nblk((long)E * 64, T), T, 0, stream>>>(h, src, dst, agg, E);
  normalize_bf16_kernel<256><<<nblk((long)N * 256, T), T, 0, stream>>>(agg, deg, aggb, N * 256);
  sage_gemm_wmma<256, 256, 256, true, true><<<MGRP * 4, 256, 0, stream>>>(
      aggb, hbB, wb2l, wb2r, h, hbA, N);

  // ---- layer 3: 256 -> 47 (padded to 64), no ReLU, straight into d_out
  zero_f32_kernel<<<nblk((long)N * 256, T), T, 0, stream>>>(agg, N * 256);
  scatter_add_kernel<256><<<nblk((long)E * 64, T), T, 0, stream>>>(h, src, dst, agg, E);
  normalize_bf16_kernel<256><<<nblk((long)N * 256, T), T, 0, stream>>>(agg, deg, aggb, N * 256);
  sage_gemm_wmma<256, 64, 47, false, false><<<MGRP, 256, 0, stream>>>(
      aggb, hbA, wb3l, wb3r, out, (unsigned short*)nullptr, N);

  // ---- log_softmax in place on d_out
  log_softmax_kernel<<<nblk(N, 8), 256, 0, stream>>>(out, N, 47);
}
